// Correlation_56745107914787
// MI455X (gfx1250) — matrix operand order repaired and checked by reading, not compiled, not run on hardware
//
#include <hip/hip_runtime.h>

typedef __attribute__((ext_vector_type(16))) _Float16 v16h;
typedef __attribute__((ext_vector_type(8)))  _Float16 v8h;
typedef __attribute__((ext_vector_type(8)))  float    v8f;

namespace {
constexpr int kBt  = 8;     // batch
constexpr int kC   = 128;   // channels (K dim of GEMM)
constexpr int kH   = 96;
constexpr int kW   = 128;
constexpr int kRad = 10;    // displacement radius in stride-2 units
constexpr int kGW  = 21;    // 2*kRad+1
constexpr int kK   = kGW * kGW;   // 441 output channels
constexpr int kHW  = kH * kW;     // 12288
constexpr int kCP  = 136;   // padded channel stride (halves): 272B rows (16B-aligned)
}

// One workgroup (8 wave32s) per (b, h). Wave (p, mt): parity p of w, M-tile mt.
// For each dy: stage f16 in2 row column-major in LDS with zero halo, run
// 4(K-chunks) x 3(N-tiles) v_wmma_f32_16x16x32_f16 per wave with double-
// buffered B fragments, branchlessly extract the |dx|<=10 band via a trash
// row, store 21x128 coalesced fp32 rows.
__global__ __launch_bounds__(256)
void corr441_wmma(const float* __restrict__ in1,
                  const float* __restrict__ in2,
                  float* __restrict__ out)
{
  // [parity][halo column u2h (0..95, data in 16..79)][channel c]
  __shared__ _Float16 Bsh[2][96][kCP];        // 51 KB
  __shared__ float    Osh[kGW + 1][kW];       // 11 KB (row kGW = write-only trash)

  const int tid  = threadIdx.x;
  const int lane = tid & 31;
  const int wv   = tid >> 5;      // 0..7
  const int p    = wv & 1;        // w parity handled by this wave
  const int mt   = wv >> 1;       // M-tile 0..3 (u = 16*mt + m)
  const int l16  = lane & 15;
  const int hi   = lane >> 4;     // half-wave id

  const int b = blockIdx.x / kH;
  const int h = blockIdx.x - b * kH;

  const size_t ibase = (size_t)b * kC * kHW;

  // ---- stage in1 row (f32 -> f16, parity split, transposed) into Bsh[..][16..79]
  {
    const float* g1 = in1 + ibase + (size_t)h * kW;
    #pragma unroll 4
    for (int i = 0; i < 32; ++i) {            // 8192 float2 / 256 threads
      int idx = tid + i * 256;
      int c = idx >> 6;                       // 64 float2 per (c) row
      int u = idx & 63;
      float2 f = *(const float2*)(g1 + (size_t)c * kHW + 2 * u);
      Bsh[0][16 + u][c] = (_Float16)f.x;
      Bsh[1][16 + u][c] = (_Float16)f.y;
    }
  }
  // ---- zero the halo columns (0..15, 80..95); never rewritten afterwards
  for (int i = tid; i < 2 * 32 * kC; i += 256) {
    int pp  = i >> 12;
    int rem = i & 4095;
    int col = rem >> 7;                       // 0..31
    int c   = rem & 127;
    Bsh[pp][(col < 16) ? col : (col + 64)][c] = (_Float16)0.f;
  }
  __syncthreads();

  // ---- pull A fragments (contiguous K 16*half + s, matching B) into registers, reused for all dy
  union AF { v16h v; v8h h2[2]; };
  AF a[4];
  {
    const _Float16* arow = &Bsh[p][16 + 16 * mt + l16][0];  // M = l16
    #pragma unroll
    for (int kk = 0; kk < 4; ++kk) {
      a[kk].h2[0] = *(const v8h*)(arow + kk * 32 + hi * 16);       // K 0..7 / 16..23
      a[kk].h2[1] = *(const v8h*)(arow + kk * 32 + hi * 16 + 8);   // K 8..15 / 24..31
    }
  }

  // three B-column base pointers for this wave's N-tiles (halo cols 16*(mt+t))
  const _Float16* bcol0 = &Bsh[p][16 * (mt + 0) + l16][0];
  const _Float16* bcol1 = &Bsh[p][16 * (mt + 1) + l16][0];
  const _Float16* bcol2 = &Bsh[p][16 * (mt + 2) + l16][0];

  const float scale = 1.0f / (float)kC;

  for (int dyi = 0; dyi < kGW; ++dyi) {
    const int y2 = h + 2 * (dyi - kRad);
    const bool valid = (y2 >= 0) && (y2 < kH);

    __syncthreads();   // previous compute done reading Bsh / stores done with Osh
    if (valid) {
      const float* g2 = in2 + ibase + (size_t)y2 * kW;
      #pragma unroll 4
      for (int i = 0; i < 32; ++i) {
        int idx = tid + i * 256;
        int c = idx >> 6;
        int u = idx & 63;
        float2 f = *(const float2*)(g2 + (size_t)c * kHW + 2 * u);
        Bsh[0][16 + u][c] = (_Float16)f.x;
        Bsh[1][16 + u][c] = (_Float16)f.y;
      }
    }
    __syncthreads();

    if (valid) {
      auto loadB = [&](int kk, AF* dst) {
        dst[0].h2[0] = *(const v8h*)(bcol0 + kk * 32 + hi * 16);   // K 0..15 / 16..31
        dst[0].h2[1] = *(const v8h*)(bcol0 + kk * 32 + hi * 16 + 8);
        dst[1].h2[0] = *(const v8h*)(bcol1 + kk * 32 + hi * 16);
        dst[1].h2[1] = *(const v8h*)(bcol1 + kk * 32 + hi * 16 + 8);
        dst[2].h2[0] = *(const v8h*)(bcol2 + kk * 32 + hi * 16);
        dst[2].h2[1] = *(const v8h*)(bcol2 + kk * 32 + hi * 16 + 8);
      };

      AF bb[2][3];
      loadB(0, bb[0]);
      v8f acc0 = {}, acc1 = {}, acc2 = {};
      #pragma unroll
      for (int kk = 0; kk < 4; ++kk) {
        if (kk < 3) loadB(kk + 1, bb[(kk + 1) & 1]);   // prefetch next K-chunk
        const AF* bt = bb[kk & 1];
        acc0 = __builtin_amdgcn_wmma_f32_16x16x32_f16(false, a[kk].v, false, bt[0].v, (short)0, acc0, false, false);
        acc1 = __builtin_amdgcn_wmma_f32_16x16x32_f16(false, a[kk].v, false, bt[1].v, (short)0, acc1, false, false);
        acc2 = __builtin_amdgcn_wmma_f32_16x16x32_f16(false, a[kk].v, false, bt[2].v, (short)0, acc2, false, false);
      }

      // ---- branchless band extraction: D vgpr r, lane l -> M = r+8*(l/16), N = l%16
      #pragma unroll
      for (int t = 0; t < 3; ++t) {
        #pragma unroll
        for (int r = 0; r < 8; ++r) {
          int m  = r + 8 * hi;
          int dx = 16 * (t - 1) + l16 - m;    // u2 - u
          bool ok = (unsigned)(dx + kRad) <= (unsigned)(2 * kRad);
          int row = ok ? (dx + kRad) : kGW;   // out-of-band -> trash row
          float v = (t == 0) ? acc0[r] : ((t == 1) ? acc1[r] : acc2[r]);
          Osh[row][32 * mt + 2 * m + p] = v * scale;
        }
      }
    }
    __syncthreads();

    // ---- coalesced store of 21 x 128 fp32 (zeros when y2 out of range)
    float* orow = out + ((size_t)b * kK + (size_t)dyi * kGW) * kHW + (size_t)h * kW;
    for (int j = 0; j < 11; ++j) {
      int idx = tid + j * 256;
      if (idx < kGW * kW) {
        int dxi = idx >> 7;
        int w   = idx & 127;
        orow[(size_t)dxi * kHW + w] = valid ? Osh[dxi][w] : 0.0f;
      }
    }
  }
}

extern "C" void kernel_launch(void* const* d_in, const int* in_sizes, int n_in,
                              void* d_out, int out_size, void* d_ws, size_t ws_size,
                              hipStream_t stream) {
  (void)in_sizes; (void)n_in; (void)d_ws; (void)ws_size; (void)out_size;
  const float* in1 = (const float*)d_in[0];
  const float* in2 = (const float*)d_in[1];
  float* out = (float*)d_out;
  corr441_wmma<<<dim3(kBt * kH), dim3(256), 0, stream>>>(in1, in2, out);
}
